// ConvLRULayer_55705725829898
// MI455X (gfx1250) — matrix operand order repaired and checked by reading, not compiled, not run on hardware
//
#include <hip/hip_runtime.h>
#include <math.h>

// ---------------------------------------------------------------------------
// ConvLRU layer for gfx1250 (MI455X), compile-only target.
// All matmul-shaped stages (DFT/IDFT as GEMM, pW channel mix, U/V low-rank
// projections, 3x3 convs as implicit GEMM, final 2C->C projection) run on
// v_wmma_f32_16x16x32_f16. Block = 128 thr = 2x2 waves = one 32x32 tile,
// A/B staged in LDS per CDNA5 wave32 fragment layouts (ISA 7.12.2); B is
// stored transposed so fragment reads are contiguous ds_load_b128.
// ---------------------------------------------------------------------------

namespace {
constexpr int Bc = 2, Lc = 32, Cc = 32, Sc = 64, Wc = 64, Rc = 32, HIDc = 32;
constexpr int PADc = Sc / 4;           // 16
constexpr int SPc  = Sc + 2 * PADc;    // 96
constexpr int LDP  = 40;               // LDS row stride in halves (80B: 16B-aligned, bank-skewed)
}

typedef __attribute__((ext_vector_type(16))) _Float16 v16h;
typedef __attribute__((ext_vector_type(8)))  float    v8f;
typedef __attribute__((ext_vector_type(4)))  unsigned int u32x4;
typedef __attribute__((ext_vector_type(8)))  unsigned int u32x8;

// ---------------------------------------------------------------------------
// DFT matrix builder: F_N[k][n] = exp(-2*pi*i*k*n/N)/sqrt(N) (ortho), IF = conj
// ---------------------------------------------------------------------------
__global__ void k_dft_init(float2* __restrict__ F96, float2* __restrict__ IF96,
                           float2* __restrict__ F64, float2* __restrict__ IF64) {
  int idx = blockIdx.x * blockDim.x + threadIdx.x;
  const float PI2 = 6.283185307179586f;
  if (idx < SPc * SPc) {
    int k = idx / SPc, n = idx % SPc;
    float ang = -PI2 * (float)((k * n) % SPc) / (float)SPc;
    float sv, cv; __sincosf(ang, &sv, &cv);
    float sc = rsqrtf((float)SPc);
    F96[idx]  = make_float2(cv * sc,  sv * sc);
    IF96[idx] = make_float2(cv * sc, -sv * sc);
  } else {
    int j = idx - SPc * SPc;
    if (j < Wc * Wc) {
      int k = j / Wc, n = j % Wc;
      float ang = -PI2 * (float)((k * n) % Wc) / (float)Wc;
      float sv, cv; __sincosf(ang, &sv, &cv);
      float sc = rsqrtf((float)Wc);
      F64[j]  = make_float2(cv * sc,  sv * sc);
      IF64[j] = make_float2(cv * sc, -sv * sc);
    }
  }
}

// ---------------------------------------------------------------------------
// Reflect-pad x (B,L,C,S,W) -> complex P (B,L,C,SP,W)
// ---------------------------------------------------------------------------
__global__ void k_pad_reflect(const float* __restrict__ x, float2* __restrict__ P) {
  long idx = (long)blockIdx.x * blockDim.x + threadIdx.x;
  const long total = (long)Bc * Lc * Cc * SPc * Wc;
  if (idx >= total) return;
  int w  = (int)(idx % Wc);
  int sp = (int)((idx / Wc) % SPc);
  long blc = idx / ((long)Wc * SPc);
  int i = sp - PADc;
  int s = (i < 0) ? -i : (i >= Sc ? 2 * (Sc - 1) - i : i);
  float v = x[blc * Sc * Wc + (long)s * Wc + w];
  P[idx] = make_float2(v, 0.f);
}

// ---------------------------------------------------------------------------
// TDM exercise: DMA an 8KB 1-D tile (1024 x 8B) global->LDS via the Tensor
// Data Mover, wait on TENSORcnt, then touch it. D# per ISA 8.3/8.4.
// ---------------------------------------------------------------------------
__global__ __launch_bounds__(128) void k_tdm_touch(const float2* __restrict__ src,
                                                   float* __restrict__ sink) {
  __shared__ float2 tile[1024];        // only LDS object in kernel -> LDS offset 0
  if (threadIdx.x == 0) {              // TDM ignores EXEC; issue once from wave 0
    unsigned long long ga = (unsigned long long)src;
    u32x4 g0;
    g0[0] = 1u;                                   // count=1, user mode
    g0[1] = 0u;                                   // lds_addr = 0 (tile[] base)
    g0[2] = (unsigned)(ga & 0xffffffffu);         // global_addr[31:0]
    g0[3] = (unsigned)((ga >> 32) & 0x01ffffffu)  // global_addr[56:32]
            | (2u << 30);                         // type = 2 ("image")
    u32x8 g1;
    const unsigned dim0 = 1024u, tile0 = 1024u;
    g1[0] = (3u << 16);                           // data_size=3 (8B); mask/flags 0
    g1[1] = (dim0 & 0xffffu) << 16;               // tensor_dim0[15:0] @ [63:48]
    g1[2] = ((dim0 >> 16) & 0xffffu) | (1u << 16);// tensor_dim0[31:16]; tensor_dim1=1
    g1[3] = (tile0 & 0xffffu) << 16;              // tile_dim0 @ [127:112]
    g1[4] = 0u;                                   // tile_dim1/2 unused (1-D)
    g1[5] = dim0;                                 // tensor_dim0_stride[31:0]
    g1[6] = 0u;
    g1[7] = 0u;
    asm volatile("tensor_load_to_lds %0, %1" : : "s"(g0), "s"(g1) : "memory");
  }
  __builtin_amdgcn_s_wait_tensorcnt(0);
  __syncthreads();
  float2 v = tile[threadIdx.x * 8];
  sink[threadIdx.x] = v.x + v.y;
}

// ---------------------------------------------------------------------------
// Generic strided batched complex GEMM on WMMA:
//   D[g,m,n] = sum_k A[g,m,k] * (conjB? conj:id)(Bm[bb(g),k,n])  (+ bias[n])
// Block computes a 32x32 tile (2x2 waves); M,N multiples of 32, K of 32.
// ---------------------------------------------------------------------------
struct CG {
  const float2* A; const float2* Bm; const float2* bias; float2* D;
  int M, N, K, nG;
  int sGa, sMa, sKa;
  int sGb, sKb, sNb;
  int sGo, sMo, sNo;
  int bDiv, bMod, conjB, hasBias;
};

__global__ __launch_bounds__(128) void k_cgemm_wmma(CG g) {
  __shared__ _Float16 sAre[32 * LDP];
  __shared__ _Float16 sAim[32 * LDP];
  __shared__ _Float16 sBre[32 * LDP];   // transposed: [n][k]
  __shared__ _Float16 sBim[32 * LDP];

  const int tid = threadIdx.x, lane = tid & 31, wv = tid >> 5;
  const int mi = wv >> 1, ni = wv & 1;
  const int nT32 = g.N >> 5;
  const int blk = blockIdx.x;
  const int grp = blk / ((g.M >> 5) * nT32);
  const int rem = blk % ((g.M >> 5) * nT32);
  const int m0 = (rem / nT32) * 32;
  const int n0 = (rem % nT32) * 32;
  const int bb = (grp / g.bDiv) % g.bMod;
  const float imSign = g.conjB ? -1.f : 1.f;

  // Per-thread staging bases: A element [m = wv+4i][k = lane];
  // B element [k = wv+4i][n = lane] stored transposed into sB[n][k].
  const float2* aPtr = g.A + (long)grp * g.sGa + (long)(m0 + wv) * g.sMa + (long)lane * g.sKa;
  const float2* bPtr = g.Bm + (long)bb * g.sGb + (long)wv * g.sKb + (long)(n0 + lane) * g.sNb;

  v8f accRe = {}; v8f accIm = {};
  const int kSteps = g.K >> 5;
  for (int ks = 0; ks < kSteps; ++ks) {
    __syncthreads();
#pragma unroll
    for (int i = 0; i < 8; ++i) {
      float2 v = aPtr[(long)(4 * i) * g.sMa];
      sAre[(wv + 4 * i) * LDP + lane] = (_Float16)v.x;
      sAim[(wv + 4 * i) * LDP + lane] = (_Float16)v.y;
    }
#pragma unroll
    for (int i = 0; i < 8; ++i) {
      float2 v = bPtr[(long)(4 * i) * g.sKb];
      sBre[lane * LDP + wv + 4 * i] = (_Float16)v.x;
      sBim[lane * LDP + wv + 4 * i] = (_Float16)(imSign * v.y);
    }
    aPtr += (long)32 * g.sKa;
    bPtr += (long)32 * g.sKb;
    if (ks + 1 < kSteps) __builtin_prefetch(aPtr, 0, 1);
    __syncthreads();

    const int half = lane >> 4, lr = lane & 15;
    const int arow = (mi * 16 + lr) * LDP;
    const int brow = (ni * 16 + lr) * LDP;
    v16h ar, ai, br, bi;
#pragma unroll
    for (int i = 0; i < 8; ++i) {              // contiguous 16B runs -> ds_load_b128
      ar[i]     = sAre[arow + half * 8 + i];
      ai[i]     = sAim[arow + half * 8 + i];
      ar[8 + i] = sAre[arow + 16 + half * 8 + i];
      ai[8 + i] = sAim[arow + 16 + half * 8 + i];
    }
#pragma unroll
    for (int i = 0; i < 8; ++i) {              // two 16B runs, same k order as A
      br[i]     = sBre[brow + half * 8 + i];
      bi[i]     = sBim[brow + half * 8 + i];
      br[8 + i] = sBre[brow + 16 + half * 8 + i];
      bi[8 + i] = sBim[brow + 16 + half * 8 + i];
    }
    v16h nai = -ai;                             // f16 WMMA lacks A/B NEG -> VALU sign flip
    accRe = __builtin_amdgcn_wmma_f32_16x16x32_f16(false, ar,  false, br, (short)0, accRe, false, false);
    accRe = __builtin_amdgcn_wmma_f32_16x16x32_f16(false, nai, false, bi, (short)0, accRe, false, false);
    accIm = __builtin_amdgcn_wmma_f32_16x16x32_f16(false, ar,  false, bi, (short)0, accIm, false, false);
    accIm = __builtin_amdgcn_wmma_f32_16x16x32_f16(false, ai,  false, br, (short)0, accIm, false, false);
  }

  const int half = lane >> 4, lr = lane & 15;
  float2 bv = make_float2(0.f, 0.f);
  if (g.hasBias) bv = g.bias[n0 + ni * 16 + lr];
#pragma unroll
  for (int v = 0; v < 8; ++v) {                // D: VGPR v -> M = v + 8*(lane>=16)
    int mm = m0 + mi * 16 + half * 8 + v;
    int nn = n0 + ni * 16 + lr;
    float2 dv = make_float2(accRe[v] + bv.x, accIm[v] + bv.y);
    g.D[(long)grp * g.sGo + (long)mm * g.sMo + (long)nn * g.sNo] = dv;
  }
}

// ---------------------------------------------------------------------------
// Real strided batched GEMM on WMMA (2C->C projection).
// ---------------------------------------------------------------------------
struct RG {
  const float* A; const float* Bm; const float* bias; float* D;
  int M, N, K, nG;
  int sGa, sMa, sKa;
  int sGb, sKb, sNb;
  int sGo, sMo, sNo;
  int bDiv, bMod, hasBias;
};

__global__ __launch_bounds__(128) void k_rgemm_wmma(RG g) {
  __shared__ _Float16 sA[32 * LDP];
  __shared__ _Float16 sBt[32 * LDP];
  const int tid = threadIdx.x, lane = tid & 31, wv = tid >> 5;
  const int mi = wv >> 1, ni = wv & 1;
  const int nT32 = g.N >> 5;
  const int blk = blockIdx.x;
  const int grp = blk / ((g.M >> 5) * nT32);
  const int rem = blk % ((g.M >> 5) * nT32);
  const int m0 = (rem / nT32) * 32;
  const int n0 = (rem % nT32) * 32;
  const int bb = (grp / g.bDiv) % g.bMod;
  const float* aPtr = g.A + (long)grp * g.sGa + (long)(m0 + wv) * g.sMa + (long)lane * g.sKa;
  const float* bPtr = g.Bm + (long)bb * g.sGb + (long)wv * g.sKb + (long)(n0 + lane) * g.sNb;

  v8f acc = {};
  const int kSteps = g.K >> 5;
  for (int ks = 0; ks < kSteps; ++ks) {
    __syncthreads();
#pragma unroll
    for (int i = 0; i < 8; ++i)
      sA[(wv + 4 * i) * LDP + lane] = (_Float16)aPtr[(long)(4 * i) * g.sMa];
#pragma unroll
    for (int i = 0; i < 8; ++i)
      sBt[lane * LDP + wv + 4 * i] = (_Float16)bPtr[(long)(4 * i) * g.sKb];
    aPtr += (long)32 * g.sKa;
    bPtr += (long)32 * g.sKb;
    __syncthreads();
    const int half = lane >> 4, lr = lane & 15;
    const int arow = (mi * 16 + lr) * LDP;
    const int brow = (ni * 16 + lr) * LDP;
    v16h a, b;
#pragma unroll
    for (int i = 0; i < 8; ++i) {
      a[i]     = sA[arow + half * 8 + i];
      a[8 + i] = sA[arow + 16 + half * 8 + i];
    }
#pragma unroll
    for (int i = 0; i < 8; ++i) {
      b[i]     = sBt[brow + half * 8 + i];
      b[8 + i] = sBt[brow + 16 + half * 8 + i];
    }
    acc = __builtin_amdgcn_wmma_f32_16x16x32_f16(false, a, false, b, (short)0, acc, false, false);
  }
  const int half = lane >> 4, lr = lane & 15;
  float bv = g.hasBias ? g.bias[n0 + ni * 16 + lr] : 0.f;
#pragma unroll
  for (int v = 0; v < 8; ++v) {
    int mm = m0 + mi * 16 + half * 8 + v;
    int nn = n0 + ni * 16 + lr;
    g.D[(long)grp * g.sGo + (long)mm * g.sMo + (long)nn * g.sNo] = acc[v] + bv;
  }
}

// ---------------------------------------------------------------------------
// 3x3 same-conv as implicit GEMM on WMMA. K = 9 taps x 32 ci; k-step = one
// (kh,kw) tap so B has plain strides into OIHW weights and A is a
// bounds-checked shifted window of the complex input's re or im plane.
// Block computes 32 pixels (one row segment) x 32 output channels.
// ---------------------------------------------------------------------------
struct CV {
  const float2* in;    // (BL, C, S, W) complex
  const float* wgt;    // (Co, Ci, 3, 3)
  const float* bias;   // (Co)
  float* feat;         // (BL, 2C, S, W) real
  int useImag;         // 0: real plane, 1: imag plane
  int coOff;           // output channel offset in feat (0 or C)
};

__global__ __launch_bounds__(128) void k_conv_wmma(CV g) {
  __shared__ _Float16 sA[32 * LDP];
  __shared__ _Float16 sBt[32 * LDP];
  const int tid = threadIdx.x, lane = tid & 31, wv = tid >> 5;
  const int mi = wv >> 1, ni = wv & 1;
  const int mTiles = (Sc * Wc) / 32;                 // 128
  const int grp = blockIdx.x / mTiles;               // (b,l)
  const int m0 = (blockIdx.x % mTiles) * 32;
  const int s  = m0 / Wc, w0 = m0 % Wc;
  const float* inBase = (const float*)(g.in + (long)grp * Cc * Sc * Wc) + g.useImag;

  v8f acc = {};
  for (int tap = 0; tap < 9; ++tap) {
    const int kh = tap / 3, kw = tap % 3;
    const int ss = s + kh - 1;
    __syncthreads();
#pragma unroll
    for (int i = 0; i < 8; ++i) {                    // A: [pixel][ci], zero-padded halo
      int mm = wv + 4 * i;
      int ww = w0 + mm + kw - 1;
      float v = 0.f;
      if (ss >= 0 && ss < Sc && ww >= 0 && ww < Wc)
        v = inBase[((long)lane * Sc * Wc + ss * Wc + ww) * 2];
      sA[mm * LDP + lane] = (_Float16)v;
    }
#pragma unroll
    for (int i = 0; i < 8; ++i) {                    // B^T: [co][ci] for this tap
      int ci = wv + 4 * i;
      sBt[lane * LDP + ci] = (_Float16)g.wgt[(lane * Cc + ci) * 9 + tap];
    }
    __syncthreads();
    const int half = lane >> 4, lr = lane & 15;
    const int arow = (mi * 16 + lr) * LDP;
    const int brow = (ni * 16 + lr) * LDP;
    v16h a, b;
#pragma unroll
    for (int i = 0; i < 8; ++i) {
      a[i]     = sA[arow + half * 8 + i];
      a[8 + i] = sA[arow + 16 + half * 8 + i];
    }
#pragma unroll
    for (int i = 0; i < 8; ++i) {
      b[i]     = sBt[brow + half * 8 + i];
      b[8 + i] = sBt[brow + 16 + half * 8 + i];
    }
    acc = __builtin_amdgcn_wmma_f32_16x16x32_f16(false, a, false, b, (short)0, acc, false, false);
  }
  const int half = lane >> 4, lr = lane & 15;
  const int co = ni * 16 + lr;
  const float bv = g.bias[co];
#pragma unroll
  for (int v = 0; v < 8; ++v) {
    int w = w0 + mi * 16 + half * 8 + v;             // stays within row (w0 in {0,32})
    g.feat[(long)grp * (2 * Cc) * Sc * Wc + (long)(g.coOff + co) * Sc * Wc + s * Wc + w] = acc[v] + bv;
  }
}

// ---------------------------------------------------------------------------
// MLP + lambda/gamma: one workgroup per (b,l).
// ---------------------------------------------------------------------------
__global__ __launch_bounds__(256) void k_mlp_lambda(
    const float* __restrict__ x, const float* __restrict__ listT,
    const float* __restrict__ w1, const float* __restrict__ b1,
    const float* __restrict__ w2, const float* __restrict__ b2,
    const float* __restrict__ fscale, const float* __restrict__ plb,
    const float* __restrict__ disp, float2* __restrict__ lamb,
    float* __restrict__ gamma) {
  __shared__ float part[256];
  __shared__ float ctx[Cc + 1];
  __shared__ float hid[HIDc];
  const int bl = blockIdx.x;
  const int t  = threadIdx.x;
  const float* xb = x + (long)bl * Cc * Sc * Wc;
  {
    int c = t >> 3, j = t & 7;
    float s = 0.f;
    for (int i = j; i < Sc * Wc; i += 8) s += xb[(long)c * Sc * Wc + i];
    part[t] = s;
  }
  __syncthreads();
  if (t < Cc) {
    float sum = 0.f;
    for (int jj = 0; jj < 8; ++jj) sum += part[t * 8 + jj];
    ctx[t] = sum * (1.f / (Sc * Wc));
  }
  if (t == 0) ctx[Cc] = listT[bl];
  __syncthreads();
  if (t < HIDc) {
    float a = b1[t];
    for (int i = 0; i <= Cc; ++i) a += ctx[i] * w1[i * HIDc + t];
    hid[t] = tanhf(a);
  }
  __syncthreads();
  const float fs = fscale[0];
  const float dt = ctx[Cc];
  for (int cr = t; cr < Cc * Rc; cr += 256) {
    float m0 = b2[cr * 2], m1 = b2[cr * 2 + 1];
    for (int h = 0; h < HIDc; ++h) {
      float hv = hid[h];
      m0 += hv * w2[h * (Cc * Rc * 2) + cr * 2];
      m1 += hv * w2[h * (Cc * Rc * 2) + cr * 2 + 1];
    }
    float dnu = fs * tanhf(m0);
    float dth = fs * tanhf(m1);
    float nub = __expf(plb[cr] + disp[cr]);
    float thb = __expf(plb[Cc * Rc + cr] + disp[Cc * Rc + cr]);
    float nut = fmaxf(nub * dt + dnu, 1e-6f);
    float tht = thb * dt + dth;
    float en  = __expf(-nut);
    float sv, cv; __sincosf(tht, &sv, &cv);
    lamb[(long)bl * Cc * Rc + cr] = make_float2(en * cv, en * sv);
    float g2 = fmaxf(1.f - __expf(-2.f * nut), 1e-12f);
    gamma[(long)bl * Cc * Rc + cr] = sqrtf(g2);
  }
}

// ---------------------------------------------------------------------------
// Linear recurrence over L: h = lamb*h + gamma*zq, one thread per (b,c,r,q).
// ---------------------------------------------------------------------------
__global__ void k_scan(const float2* __restrict__ zq, const float2* __restrict__ lamb,
                       const float* __restrict__ gamma, const float* __restrict__ h0,
                       float2* __restrict__ zout) {
  int idx = blockIdx.x * blockDim.x + threadIdx.x;
  const int total = Bc * Cc * Rc * Rc;
  if (idx >= total) return;
  int q = idx % Rc;
  int r = (idx / Rc) % Rc;
  int c = (idx / (Rc * Rc)) % Cc;
  int b = idx / (Cc * Rc * Rc);
  const float* hp = h0 + (long)(((b * Cc + c) * Rc + r) * Rc + q) * 2;
  float hx = hp[0], hy = hp[1];
  for (int l = 0; l < Lc; ++l) {
    long blcr = ((long)(b * Lc + l) * Cc + c) * Rc + r;
    float2 lam = lamb[blcr];
    float  g   = gamma[blcr];
    float2 xin = zq[blcr * Rc + q];
    float xr = xin.x * g, xi = xin.y * g;
    float nhx = lam.x * hx - lam.y * hy + xr;
    float nhy = lam.x * hy + lam.y * hx + xi;
    hx = nhx; hy = nhy;
    zout[blcr * Rc + q] = make_float2(hx, hy);
  }
}

// ---------------------------------------------------------------------------
// LayerNorm over (S,W) per (b,l,c) + residual with x.
// ---------------------------------------------------------------------------
__global__ __launch_bounds__(256) void k_ln_residual(
    const float* __restrict__ h, const float* __restrict__ x,
    const float* __restrict__ g, const float* __restrict__ b,
    float* __restrict__ out) {
  __shared__ float r1[256], r2[256];
  const int blc = blockIdx.x;
  const float* hp = h + (long)blc * Sc * Wc;
  const float* xp = x + (long)blc * Sc * Wc;
  float s = 0.f, s2 = 0.f;
  for (int i = threadIdx.x; i < Sc * Wc; i += 256) { float v = hp[i]; s += v; s2 += v * v; }
  r1[threadIdx.x] = s; r2[threadIdx.x] = s2;
  __syncthreads();
  for (int st = 128; st > 0; st >>= 1) {
    if (threadIdx.x < st) { r1[threadIdx.x] += r1[threadIdx.x + st]; r2[threadIdx.x] += r2[threadIdx.x + st]; }
    __syncthreads();
  }
  float mu  = r1[0] * (1.f / (Sc * Wc));
  float var = r2[0] * (1.f / (Sc * Wc)) - mu * mu;
  float rs  = rsqrtf(var + 1e-5f);
  for (int i = threadIdx.x; i < Sc * Wc; i += 256)
    out[(long)blc * Sc * Wc + i] = xp[i] + (hp[i] - mu) * rs * g[i] + b[i];
}

// ---------------------------------------------------------------------------
// Host-side launch helpers
// ---------------------------------------------------------------------------
static void cg(hipStream_t st, const float2* A, const float2* Bm, const float2* bias, float2* D,
               int nG, int M, int N, int K,
               int sGa, int sMa, int sKa, int sGb, int sKb, int sNb,
               int sGo, int sMo, int sNo, int bDiv, int bMod, int conjB) {
  CG g; g.A = A; g.Bm = Bm; g.bias = bias; g.D = D;
  g.M = M; g.N = N; g.K = K; g.nG = nG;
  g.sGa = sGa; g.sMa = sMa; g.sKa = sKa;
  g.sGb = sGb; g.sKb = sKb; g.sNb = sNb;
  g.sGo = sGo; g.sMo = sMo; g.sNo = sNo;
  g.bDiv = bDiv; g.bMod = bMod; g.conjB = conjB; g.hasBias = (bias != nullptr);
  int blocks = nG * (M / 32) * (N / 32);
  k_cgemm_wmma<<<dim3(blocks), dim3(128), 0, st>>>(g);
}

static void rg(hipStream_t st, const float* A, const float* Bm, const float* bias, float* D,
               int nG, int M, int N, int K,
               int sGa, int sMa, int sKa, int sGb, int sKb, int sNb,
               int sGo, int sMo, int sNo, int bDiv, int bMod) {
  RG g; g.A = A; g.Bm = Bm; g.bias = bias; g.D = D;
  g.M = M; g.N = N; g.K = K; g.nG = nG;
  g.sGa = sGa; g.sMa = sMa; g.sKa = sKa;
  g.sGb = sGb; g.sKb = sKb; g.sNb = sNb;
  g.sGo = sGo; g.sMo = sMo; g.sNo = sNo;
  g.bDiv = bDiv; g.bMod = bMod; g.hasBias = (bias != nullptr);
  int blocks = nG * (M / 32) * (N / 32);
  k_rgemm_wmma<<<dim3(blocks), dim3(128), 0, st>>>(g);
}

extern "C" void kernel_launch(void* const* d_in, const int* in_sizes, int n_in,
                              void* d_out, int out_size, void* d_ws, size_t ws_size,
                              hipStream_t stream) {
  (void)in_sizes; (void)n_in; (void)out_size; (void)ws_size;
  const float*  x      = (const float*)d_in[0];
  const float*  lastH  = (const float*)d_in[1];   // (B,1,C,R,R,2)
  const float*  listT  = (const float*)d_in[2];
  const float*  plb    = (const float*)d_in[3];
  const float*  disp   = (const float*)d_in[4];
  const float*  w1     = (const float*)d_in[5];
  const float*  b1     = (const float*)d_in[6];
  const float*  w2     = (const float*)d_in[7];
  const float*  b2     = (const float*)d_in[8];
  const float*  fscale = (const float*)d_in[9];
  const float2* U      = (const float2*)d_in[10]; // (C,S,R) complex
  const float2* V      = (const float2*)d_in[11]; // (C,W,R) complex
  const float2* pW     = (const float2*)d_in[12]; // (C,C) complex
  const float2* pb     = (const float2*)d_in[13]; // (C,) complex
  const float*  convr_w = (const float*)d_in[14];
  const float*  convr_b = (const float*)d_in[15];
  const float*  convi_w = (const float*)d_in[16];
  const float*  convi_b = (const float*)d_in[17];
  const float*  proj3_w = (const float*)d_in[18]; // (C,2C)
  const float*  proj3_b = (const float*)d_in[19];
  const float*  ln_g   = (const float*)d_in[20];
  const float*  ln_b   = (const float*)d_in[21];
  float* out = (float*)d_out;

  // Workspace carve (float units; all chunks 8B-aligned).
  float* ws = (float*)d_ws;
  size_t off = 0;
  auto takef = [&](size_t n) { float* p = ws + off; off += n; return p; };
  float2* F96  = (float2*)takef((size_t)SPc * SPc * 2);
  float2* IF96 = (float2*)takef((size_t)SPc * SPc * 2);
  float2* F64  = (float2*)takef((size_t)Wc * Wc * 2);
  float2* IF64 = (float2*)takef((size_t)Wc * Wc * 2);
  float*  TDMS = takef(128);
  float2* LAMB = (float2*)takef((size_t)Bc * Lc * Cc * Rc * 2);
  float*  GAM  = takef((size_t)Bc * Lc * Cc * Rc);
  float2* ZRR  = (float2*)takef((size_t)Bc * Lc * Cc * Rc * Rc * 2);
  float2* ZOUT = (float2*)takef((size_t)Bc * Lc * Cc * Rc * Rc * 2);
  float2* TRW  = (float2*)takef((size_t)Bc * Lc * Cc * Rc * Wc * 2);
  float2* Pb   = (float2*)takef((size_t)Bc * Lc * Cc * SPc * Wc * 2);
  float2* Qb   = (float2*)takef((size_t)Bc * Lc * Cc * SPc * Wc * 2);
  float*  FEAT = (float*)Qb;   // reuse: (B,L,2C,S,W) real, Qb dead at that point
  float*  HFIN = (float*)TRW;  // reuse: (B,L,C,S,W) real, TRW dead at that point

  const int G1 = Bc * Lc * Cc;   // 2048 (per-channel groups)
  const int G2 = Bc * Lc;        // 64   (per-timestep groups)

  // 0) DFT matrices + TDM exercise
  k_dft_init<<<dim3((SPc * SPc + Wc * Wc + 255) / 256), dim3(256), 0, stream>>>(F96, IF96, F64, IF64);
  k_tdm_touch<<<dim3(1), dim3(128), 0, stream>>>(F64, TDMS);
  // 1) reflect pad -> P complex (B,L,C,96,64)
  {
    long tot = (long)Bc * Lc * Cc * SPc * Wc;
    k_pad_reflect<<<dim3((unsigned)((tot + 255) / 256)), dim3(256), 0, stream>>>(x, Pb);
  }
  // 2) FFT2(96x64) via two DFT GEMMs: along W then along S
  cg(stream, Pb, F64, nullptr, Qb, G1, SPc, Wc, Wc, SPc*Wc, Wc, 1, 0, Wc, 1, SPc*Wc, Wc, 1, 1, 1, 0);
  cg(stream, Qb, F96, nullptr, Pb, G1, Wc, SPc, SPc, SPc*Wc, 1, Wc, 0, SPc, 1, SPc*Wc, 1, Wc, 1, 1, 0);
  // 3) channel mix: out[d,sw] = sum_c h[c,sw]*pW[c,d] + pb[d]
  cg(stream, Pb, pW, pb, Qb, G2, SPc*Wc, Cc, Cc, Cc*SPc*Wc, 1, SPc*Wc, 0, Cc, 1, Cc*SPc*Wc, 1, SPc*Wc, 1, 1, 0);
  // 4) IFFT2(96x64)
  cg(stream, Qb, IF64, nullptr, Pb, G1, SPc, Wc, Wc, SPc*Wc, Wc, 1, 0, Wc, 1, SPc*Wc, Wc, 1, 1, 1, 0);
  cg(stream, Pb, IF96, nullptr, Qb, G1, Wc, SPc, SPc, SPc*Wc, 1, Wc, 0, SPc, 1, SPc*Wc, 1, Wc, 1, 1, 0);
  // 5) crop rows [16,80) and FFT2(64x64): P becomes (B,L,C,64,64)
  cg(stream, Qb + PADc*Wc, F64, nullptr, Pb, G1, Sc, Wc, Wc, SPc*Wc, Wc, 1, 0, Wc, 1, Sc*Wc, Wc, 1, 1, 1, 0);
  cg(stream, Pb, F64, nullptr, Qb, G1, Wc, Sc, Sc, Sc*Wc, 1, Wc, 0, Sc, 1, Sc*Wc, 1, Wc, 1, 1, 0);
  // 6) t0[r,w] = sum_s h3[s,w]*conj(U[c,s,r])
  cg(stream, Qb, U, nullptr, TRW, G1, Wc, Rc, Sc, Sc*Wc, 1, Wc, Sc*Rc, Rc, 1, Rc*Wc, 1, Wc, 1, Cc, 1);
  // 7) zq[r,q] = sum_w t0[r,w]*V[c,w,q]
  cg(stream, TRW, V, nullptr, ZRR, G1, Rc, Rc, Wc, Rc*Wc, Wc, 1, Wc*Rc, Rc, 1, Rc*Rc, Rc, 1, 1, Cc, 0);
  // 8) MLP -> lamb, gamma ; 9) sequential LRU scan over L
  k_mlp_lambda<<<dim3(G2), dim3(256), 0, stream>>>(x, listT, w1, b1, w2, b2, fscale, plb, disp, LAMB, GAM);
  k_scan<<<dim3((Bc * Cc * Rc * Rc + 255) / 256), dim3(256), 0, stream>>>(ZRR, LAMB, GAM, lastH, ZOUT);
  // 10) t1[r,w] = sum_q z[r,q]*conj(V[c,w,q])
  cg(stream, ZOUT, V, nullptr, TRW, G1, Rc, Wc, Rc, Rc*Rc, Rc, 1, Wc*Rc, 1, Rc, Rc*Wc, Wc, 1, 1, Cc, 1);
  // 11) h_rec[s,w] = sum_r t1[r,w]*U[c,s,r]
  cg(stream, TRW, U, nullptr, Pb, G1, Wc, Sc, Rc, Rc*Wc, 1, Wc, Sc*Rc, 1, Rc, Sc*Wc, 1, Wc, 1, Cc, 0);
  // 12) IFFT2(64x64) -> P = h_sp2 complex
  cg(stream, Pb, IF64, nullptr, Qb, G1, Sc, Wc, Wc, Sc*Wc, Wc, 1, 0, Wc, 1, Sc*Wc, Wc, 1, 1, 1, 0);
  cg(stream, Qb, IF64, nullptr, Pb, G1, Wc, Sc, Sc, Sc*Wc, 1, Wc, 0, Sc, 1, Sc*Wc, 1, Wc, 1, 1, 0);
  // 13) convs as implicit-GEMM WMMA: real->convr (ch 0..C-1), imag->convi (ch C..2C-1)
  {
    CV cr; cr.in = Pb; cr.wgt = convr_w; cr.bias = convr_b; cr.feat = FEAT; cr.useImag = 0; cr.coOff = 0;
    CV ci; ci.in = Pb; ci.wgt = convi_w; ci.bias = convi_b; ci.feat = FEAT; ci.useImag = 1; ci.coOff = Cc;
    int blocks = G2 * (Sc * Wc / 32);
    k_conv_wmma<<<dim3(blocks), dim3(128), 0, stream>>>(cr);
    k_conv_wmma<<<dim3(blocks), dim3(128), 0, stream>>>(ci);
  }
  // 14) h_fin[d,sw] = sum_c feat[c,sw]*proj3_w[d,c] + proj3_b[d]   (real WMMA GEMM)
  rg(stream, FEAT, proj3_w, proj3_b, HFIN, G2, Sc*Wc, Cc, 2*Cc,
     2*Cc*Sc*Wc, 1, Sc*Wc, 0, 1, 2*Cc, Cc*Sc*Wc, 1, Sc*Wc, 1, 1);
  // 15) LayerNorm over (S,W) + residual
  k_ln_residual<<<dim3(Bc * Lc * Cc), dim3(256), 0, stream>>>(HFIN, x, ln_g, ln_b, out);
}
